// QGINConv_25649544692297
// MI455X (gfx1250) — matrix operand order repaired and checked by reading, not compiled, not run on hardware
//
#include <hip/hip_runtime.h>
#include <hip/hip_bf16.h>

typedef __attribute__((ext_vector_type(16))) _Float16 v16h;
typedef __attribute__((ext_vector_type(8)))  float    v8f;

#define D_IN  64
#define E_W   16
#define D_MSG 80   // D_IN + E_W
#define D_PAD 96   // D_MSG rounded up to 3 * 32 (WMMA K-chunks)

// ---------------------------------------------------------------------------
// Kernel 1: acc[i][0:64] = feat[i][:]  (== (1+eps)*feat with eps=0)
//           acc[i][64:96] = 0          (cols 80..95 are permanent zero pad)
// ---------------------------------------------------------------------------
__global__ void gin_init_kernel(const float* __restrict__ feat,
                                float* __restrict__ acc,
                                int total /* nNodes * 96 */) {
  int i = blockIdx.x * blockDim.x + threadIdx.x;
  if (i >= total) return;
  int node = i / D_PAD;
  int j    = i - node * D_PAD;
  acc[i] = (j < D_IN) ? feat[(size_t)node * D_IN + j] : 0.0f;
}

// ---------------------------------------------------------------------------
// Kernel 1b: Wt[n][k] = (k < 80) ? W[k][n] : 0   -- transposed + zero-padded
// weights so WMMA B-operand loads are contiguous per lane. 64x96 = 24KB.
// ---------------------------------------------------------------------------
__global__ void gin_wt_kernel(const float* __restrict__ W,
                              float* __restrict__ Wt) {
  int i = blockIdx.x * blockDim.x + threadIdx.x;   // over 64*96
  if (i >= D_IN * D_PAD) return;
  int n = i / D_PAD;
  int k = i - n * D_PAD;
  Wt[i] = (k < D_MSG) ? W[(size_t)k * D_IN + n] : 0.0f;
}

// ---------------------------------------------------------------------------
// Kernel 2: wave32-per-edge scatter-add into the accumulator.
//   acc[dst[e]][0:64]  += feat[src[e]][:]
//   acc[dst[e]][64:80] += edge_w[e][:]
// 128M hw float atomics; accumulator (38MB) + feat (26MB) are L2-resident.
// ---------------------------------------------------------------------------
__global__ void gin_scatter_kernel(const float* __restrict__ feat,
                                   const float* __restrict__ edge_w,
                                   const int* __restrict__ src,
                                   const int* __restrict__ dst,
                                   float* __restrict__ acc,
                                   int nEdges) {
  int gid  = blockIdx.x * blockDim.x + threadIdx.x;
  int e    = gid >> 5;          // one wave32 per edge
  int lane = threadIdx.x & 31;
  if (e >= nEdges) return;

  int s = src[e];
  int d = dst[e];
  float*       row = acc  + (size_t)d * D_PAD;
  const float* fr  = feat + (size_t)s * D_IN;

  // 64 feature elements: 2 rounds of 32 lanes -> global_atomic_add_f32
  unsafeAtomicAdd(row + lane,      fr[lane]);
  unsafeAtomicAdd(row + lane + 32, fr[lane + 32]);
  // 16 edge-weight elements
  if (lane < E_W) {
    unsafeAtomicAdd(row + D_IN + lane, edge_w[(size_t)e * E_W + lane]);
  }
}

// ---------------------------------------------------------------------------
// Kernel 3: out = relu(acc[N,96] @ Wt^T + b) via v_wmma_f32_16x16x32_f16.
// Block = 128 threads = 4 waves; block m owns rows 16m..16m+15; wave w owns
// output columns 16w..16w+15. K = 96 = 3 chunks of 32, no predicates.
//
// Operand layouts (CDNA5 ISA):
//   A (16x32 f16): lane = M + 16*hi, half idx -> K = 16*(idx/8) + 8*hi + idx%8
//   B (32x16 f16): lane = N + 16*hi, half idx -> K = 16*(idx/8) + 8*hi + idx%8 (same as A)
//   C/D (v8f):     VGPR r holds (M = r + 8*hi, N = lane%16)
// ---------------------------------------------------------------------------
__global__ void gin_gemm_kernel(const float* __restrict__ acc,
                                const float* __restrict__ Wt,
                                const float* __restrict__ bias,
                                float* __restrict__ out) {
  const int lane  = threadIdx.x & 31;
  const int wave  = threadIdx.x >> 5;      // 0..3 -> N tile
  const int mBase = blockIdx.x * 16;
  const int nBase = wave * 16;

  const int sub = lane & 15;               // M index for A rows, N index for B/C/D
  const int hi  = lane >> 4;               // lane half select

  const float* arow = acc + (size_t)(mBase + sub) * D_PAD;   // A row for this lane
  const float* brow = Wt  + (size_t)(nBase + sub) * D_PAD;   // B column (transposed)

  v8f c = {};

  #pragma unroll
  for (int kc = 0; kc < 3; ++kc) {
    const int kBase = kc * 32;
    v16h a, b;
    // A: two groups of 8 contiguous floats each -> 2x global_load_b128 pairs
    #pragma unroll
    for (int g = 0; g < 2; ++g) {
      #pragma unroll
      for (int kk = 0; kk < 8; ++kk) {
        a[g * 8 + kk] = (_Float16)arow[kBase + g * 16 + (hi << 3) + kk];
      }
    }
    // B: two groups of 8 contiguous floats each (same K order as A)
    #pragma unroll
    for (int idx = 0; idx < 16; ++idx) {
      b[idx] = (_Float16)brow[kBase + ((idx >> 3) << 4) + (hi << 3) + (idx & 7)];
    }
    c = __builtin_amdgcn_wmma_f32_16x16x32_f16(
        /*neg_a=*/false, a, /*neg_b=*/false, b,
        /*c_mod=*/(short)0, c, /*reuse_a=*/false, /*reuse_b=*/false);
  }

  // D layout: VGPR r holds element (M = r + 8*hi, N = sub)
  const int   N  = nBase + sub;
  const float bN = bias[N];
  #pragma unroll
  for (int r = 0; r < 8; ++r) {
    int M = r + (hi << 3);
    float v = c[r] + bN;
    out[(size_t)(mBase + M) * D_IN + N] = fmaxf(v, 0.0f);
  }
}

// ---------------------------------------------------------------------------
// Launch
// ---------------------------------------------------------------------------
extern "C" void kernel_launch(void* const* d_in, const int* in_sizes, int n_in,
                              void* d_out, int out_size, void* d_ws, size_t ws_size,
                              hipStream_t stream) {
  const float* feat   = (const float*)d_in[0];  // [N, 64]
  const float* edge_w = (const float*)d_in[1];  // [E, 16]
  const float* W      = (const float*)d_in[2];  // [80, 64]
  const float* b      = (const float*)d_in[3];  // [64]
  const int*   src    = (const int*)d_in[4];    // [E]
  const int*   dst    = (const int*)d_in[5];    // [E]
  float*       out    = (float*)d_out;          // [N, 64]

  const int nNodes = in_sizes[0] / D_IN;
  const int nEdges = in_sizes[4];

  float* acc = (float*)d_ws;                          // [N, 96]  (38.4 MB)
  float* Wt  = acc + (size_t)nNodes * D_PAD;          // [64, 96] (24 KB)

  // 1) init accumulator = [feat | 0pad]
  {
    int total  = nNodes * D_PAD;
    int blocks = (total + 255) / 256;
    gin_init_kernel<<<blocks, 256, 0, stream>>>(feat, acc, total);
  }
  // 1b) build transposed + padded weights
  {
    int total  = D_IN * D_PAD;
    int blocks = (total + 255) / 256;
    gin_wt_kernel<<<blocks, 256, 0, stream>>>(W, Wt);
  }
  // 2) scatter-add edges (wave32 per edge, 8 edges per 256-thread block)
  {
    long long threads = (long long)nEdges * 32;
    int blocks = (int)((threads + 255) / 256);
    gin_scatter_kernel<<<blocks, 256, 0, stream>>>(feat, edge_w, src, dst, acc, nEdges);
  }
  // 3) WMMA GEMM + bias + relu (nNodes == 6250 * 16, exact M tiling)
  {
    int mTiles = (nNodes + 15) / 16;
    gin_gemm_kernel<<<mTiles, 128, 0, stream>>>(acc, Wt, b, out);
  }
}
